// MultiModalBlock_18717467475982
// MI455X (gfx1250) — matrix operand order repaired and checked by reading, not compiled, not run on hardware
//
#include <hip/hip_runtime.h>
#include <hip/hip_bf16.h>

// ---------------- types ----------------
typedef __bf16 bf16_t;
typedef __attribute__((ext_vector_type(16))) __bf16 v16bf;
typedef __attribute__((ext_vector_type(8)))  __bf16 v8bf;
typedef __attribute__((ext_vector_type(8)))  float  v8f;
typedef int v4i_gcc __attribute__((vector_size(16)));   // matches builtin param type

#define B_DIM 16
#define C_DIM 1024
#define H_DIM 48
#define W_DIM 48
#define N_DIM 2304      // H*W
#define L_DIM 512
#define CT_DIM 768
#define TSTR 40         // LDS tile stride (bf16 elems), 80B rows -> 16B aligned
#define PSTR 520        // probs LDS stride

// ---------------- CDNA5 async direct-to-LDS copy (16B), guarded ----------------
#if __has_builtin(__builtin_amdgcn_global_load_async_to_lds_b128)
#define HAVE_ASYNC_LDS 1
static __device__ inline void ldsdma16(const bf16_t* g, bf16_t* l) {
  __builtin_amdgcn_global_load_async_to_lds_b128(
      (__attribute__((address_space(1))) v4i_gcc*)g,
      (__attribute__((address_space(3))) v4i_gcc*)l, 0, 0);
}
static __device__ inline void lds_dma_wait() {
#if __has_builtin(__builtin_amdgcn_s_wait_asynccnt)
  __builtin_amdgcn_s_wait_asynccnt(0);
#else
  asm volatile("s_wait_asynccnt 0x0" ::: "memory");
#endif
}
#else
#define HAVE_ASYNC_LDS 0
static __device__ inline void ldsdma16(const bf16_t* g, bf16_t* l) {
  *(uint4*)l = *(const uint4*)g;
}
static __device__ inline void lds_dma_wait() {}
#endif

static __device__ inline v8f zero_v8f() {
  v8f z;
#pragma unroll
  for (int i = 0; i < 8; ++i) z[i] = 0.0f;
  return z;
}

static __device__ inline v16bf cat8(v8bf a, v8bf b) {
  v16bf r;
#pragma unroll
  for (int i = 0; i < 8; ++i) { r[i] = a[i]; r[i + 8] = b[i]; }
  return r;
}

// ---------------- fp32 -> bf16 convert ----------------
__global__ void cvt_bf16_kernel(const float* __restrict__ s, bf16_t* __restrict__ d, size_t n) {
  size_t i = (size_t)blockIdx.x * blockDim.x + threadIdx.x;
  size_t stride = (size_t)gridDim.x * blockDim.x;
  for (; i < n; i += stride) d[i] = (bf16_t)s[i];
}

// ---------------- out init: [v_feat ; t_tok] ----------------
__global__ void init_out_kernel(const float* __restrict__ vfeat, const float* __restrict__ ttok,
                                float* __restrict__ out) {
  const size_t n1 = (size_t)B_DIM * C_DIM * N_DIM;
  const size_t n2 = (size_t)B_DIM * L_DIM * CT_DIM;
  size_t i = (size_t)blockIdx.x * blockDim.x + threadIdx.x;
  size_t stride = (size_t)gridDim.x * blockDim.x;
  for (; i < n1 + n2; i += stride)
    out[i] = (i < n1) ? vfeat[i] : ttok[i - n1];
}

// ---------------- avg-pool (count_include_pad) -> (B,N,C) bf16 ----------------
__global__ __launch_bounds__(256) void pool_kernel(const float* __restrict__ vfeat,
                                                   bf16_t* __restrict__ vflat, int ks) {
  __shared__ float plane[N_DIM];
  const int t = threadIdx.x;
  const int bc = blockIdx.x;            // b*C + c
  const int b = bc >> 10;
  const int c = bc & 1023;
  const float* src = vfeat + (size_t)bc * N_DIM;
  for (int i = t; i < N_DIM; i += 256) plane[i] = src[i];
  __syncthreads();
  const int pad = ks >> 1;
  const float inv = 1.0f / (float)(ks * ks);
  for (int p = t; p < N_DIM; p += 256) {
    const int h = p / W_DIM, w = p % W_DIM;
    float s = 0.0f;
    for (int dh = -pad; dh <= pad; ++dh) {
      const int hh = h + dh;
      if (hh < 0 || hh >= H_DIM) continue;
      for (int dw = -pad; dw <= pad; ++dw) {
        const int ww = w + dw;
        if (ww < 0 || ww >= W_DIM) continue;
        s += plane[hh * W_DIM + ww];
      }
    }
    vflat[((size_t)(b * N_DIM + p)) * C_DIM + c] = (bf16_t)(s * inv);
  }
}

// ---------------- D = A(MxK) @ B(KxN) + bias, bf16 in/out, f32 WMMA accum ----------------
// Block: 256 threads (8 waves, 2x4), tile 128x128, wave tile 64x32, K-step 32.
__global__ __launch_bounds__(256) void gemm_bias_bf16(const bf16_t* __restrict__ A,
                                                      const bf16_t* __restrict__ Bw,
                                                      const float* __restrict__ bias,
                                                      bf16_t* __restrict__ D,
                                                      int M, int Nc, int K) {
  __shared__ bf16_t As[128 * TSTR];
  __shared__ bf16_t Bs[128 * TSTR];   // transposed: Bs[n][k]
  const int t = threadIdx.x;
  const int mbase = blockIdx.y * 128;
  const int nbase = blockIdx.x * 128;
  const int wid = t >> 5, lane = t & 31;
  const int wm = wid >> 2, wn = wid & 3;
  const int lr = lane & 15, hi = lane >> 4;

  v8f acc[4][2];
#pragma unroll
  for (int m = 0; m < 4; ++m)
#pragma unroll
    for (int n = 0; n < 2; ++n) acc[m][n] = zero_v8f();

  for (int k0 = 0; k0 < K; k0 += 32) {
    __syncthreads();
    // stage A: 128x32, straight copy -> async LDS-DMA (no VGPR round-trip)
#pragma unroll
    for (int j = 0; j < 2; ++j) {
      const int i = t + 256 * j;
      const int row = i >> 2, kk8 = (i & 3) << 3;
      ldsdma16(A + (size_t)(mbase + row) * K + k0 + kk8, As + row * TSTR + kk8);
    }
    // stage B transposed: 32(k) x 128(n) -> Bs[n][k]; pipeline both loads first
    {
      uint4 dv[2];
#pragma unroll
      for (int j = 0; j < 2; ++j) {
        const int i = t + 256 * j;
        const int k = i >> 4, n8 = (i & 15) << 3;
        dv[j] = *(const uint4*)(Bw + (size_t)(k0 + k) * Nc + nbase + n8);
      }
#pragma unroll
      for (int j = 0; j < 2; ++j) {
        const int i = t + 256 * j;
        const int k = i >> 4, n8 = (i & 15) << 3;
        const bf16_t* e = (const bf16_t*)&dv[j];
#pragma unroll
        for (int x = 0; x < 8; ++x) Bs[(n8 + x) * TSTR + k] = e[x];
      }
    }
    if (k0 + 32 < K) {  // pull next tiles toward L2/L0 (global_prefetch_b8)
      __builtin_prefetch(A + (size_t)(mbase + (t >> 1)) * K + (k0 + 32) + (t & 1) * 16, 0, 1);
      __builtin_prefetch(Bw + (size_t)(k0 + 32 + (t >> 4)) * Nc + nbase + (t & 15) * 8, 0, 1);
    }
    lds_dma_wait();
    __syncthreads();
    const int akb = hi ? 8 : 0;    // 16-bit A 16x32 lane layout
    const int bkb = hi ? 8 : 0;    // 16-bit B 32x16 lane layout
    v16bf af[4], bfg[2];
#pragma unroll
    for (int m = 0; m < 4; ++m) {
      const int arow = wm * 64 + m * 16 + lr;
      af[m] = cat8(*(const v8bf*)(As + arow * TSTR + akb),
                   *(const v8bf*)(As + arow * TSTR + akb + 16));
    }
#pragma unroll
    for (int n = 0; n < 2; ++n) {
      const int col = wn * 32 + n * 16 + lr;
      bfg[n] = cat8(*(const v8bf*)(Bs + col * TSTR + bkb),
                    *(const v8bf*)(Bs + col * TSTR + bkb + 16));
    }
#pragma unroll
    for (int m = 0; m < 4; ++m)
#pragma unroll
      for (int n = 0; n < 2; ++n)
        acc[m][n] = __builtin_amdgcn_wmma_f32_16x16x32_bf16(
            false, af[m], false, bfg[n], (short)0, acc[m][n], false, false);
  }
  // epilogue: + bias, cast bf16
#pragma unroll
  for (int m = 0; m < 4; ++m)
#pragma unroll
    for (int n = 0; n < 2; ++n) {
      const int gc = nbase + wn * 32 + n * 16 + lr;
      const float bv = bias[gc];
#pragma unroll
      for (int v = 0; v < 8; ++v) {
        const int gm = mbase + wm * 64 + m * 16 + v + 8 * hi;
        D[(size_t)gm * Nc + gc] = (bf16_t)(acc[m][n][v] + bv);
      }
    }
}

// ---------------- fused attention: logits -> softmax -> P @ v_txt, += into d_out ----------------
// One block = one (b, 32 q-rows). 8 waves (2x4): wave tile 16x32.
__global__ __launch_bounds__(256) void attn_kernel(const bf16_t* __restrict__ q,
                                                   const bf16_t* __restrict__ ktxt,
                                                   const bf16_t* __restrict__ vtxt,
                                                   const float* __restrict__ lam,
                                                   float* __restrict__ out, int scale_idx) {
  __shared__ bf16_t P[32 * PSTR];      // 33.3 KB probs/logits
  __shared__ bf16_t As[32 * TSTR];     // q staging
  __shared__ bf16_t Bs[128 * TSTR];    // k/v staging
  __shared__ float rmax[32], rsum[32];

  const int t = threadIdx.x;
  const int b = blockIdx.y;
  const int nbase = blockIdx.x * 32;
  const int wid = t >> 5, lane = t & 31;
  const int wm = wid >> 2, wn = wid & 3;
  const int lr = lane & 15, hi = lane >> 4;
  const float qscale = 0.03125f;       // C^-0.5 = 1/32

  // ---- pass 1: logits = q @ k_txt^T * scale ----
  for (int lc = 0; lc < 4; ++lc) {
    v8f acc0 = zero_v8f(), acc1 = zero_v8f();
    const int lbase = lc * 128;
    for (int k0 = 0; k0 < C_DIM; k0 += 32) {
      __syncthreads();
      if (t < 128) {  // q tile 32x32, async LDS-DMA
        const int row = t >> 2, kk8 = (t & 3) << 3;
        ldsdma16(q + (size_t)(b * N_DIM + nbase + row) * C_DIM + k0 + kk8,
                 As + row * TSTR + kk8);
      }
#pragma unroll
      for (int j = 0; j < 2; ++j) {  // k_txt tile: 128(l) x 32(c), contiguous -> async LDS-DMA
        const int i = t + 256 * j;
        const int l = i >> 2, kk8 = (i & 3) << 3;
        ldsdma16(ktxt + (size_t)(b * L_DIM + lbase + l) * C_DIM + k0 + kk8,
                 Bs + l * TSTR + kk8);
      }
      lds_dma_wait();
      __syncthreads();
      const int arow = wm * 16 + lr;
      const int akb = hi ? 8 : 0;
      const v16bf af = cat8(*(const v8bf*)(As + arow * TSTR + akb),
                            *(const v8bf*)(As + arow * TSTR + akb + 16));
      const int bkb = hi ? 8 : 0;
      const int col0 = wn * 32 + lr, col1 = col0 + 16;
      const v16bf bf0 = cat8(*(const v8bf*)(Bs + col0 * TSTR + bkb),
                             *(const v8bf*)(Bs + col0 * TSTR + bkb + 16));
      const v16bf bf1 = cat8(*(const v8bf*)(Bs + col1 * TSTR + bkb),
                             *(const v8bf*)(Bs + col1 * TSTR + bkb + 16));
      acc0 = __builtin_amdgcn_wmma_f32_16x16x32_bf16(false, af, false, bf0, (short)0, acc0, false, false);
      acc1 = __builtin_amdgcn_wmma_f32_16x16x32_bf16(false, af, false, bf1, (short)0, acc1, false, false);
    }
    __syncthreads();
#pragma unroll
    for (int v = 0; v < 8; ++v) {
      const int row = wm * 16 + v + 8 * hi;
      const int c0 = lbase + wn * 32 + lr;
      P[row * PSTR + c0]      = (bf16_t)(acc0[v] * qscale);
      P[row * PSTR + c0 + 16] = (bf16_t)(acc1[v] * qscale);
    }
  }
  __syncthreads();

  // ---- softmax over L=512, in-place in P ----
  if (t < 32) {
    float mx = -1e30f;
    for (int c = 0; c < L_DIM; ++c) mx = fmaxf(mx, (float)P[t * PSTR + c]);
    float s = 0.0f;
    for (int c = 0; c < L_DIM; ++c) s += __expf((float)P[t * PSTR + c] - mx);
    rmax[t] = mx;
    rsum[t] = 1.0f / s;
  }
  __syncthreads();
  for (int j = 0; j < 64; ++j) {
    const int i = t + 256 * j;
    const int r = i >> 9, c = i & 511;
    P[r * PSTR + c] = (bf16_t)(__expf((float)P[r * PSTR + c] - rmax[r]) * rsum[r]);
  }
  __syncthreads();

  // scale-mix weight = softmax(lam)[scale_idx]
  const float l0 = lam[0], l1 = lam[1], l2 = lam[2];
  const float lm = fmaxf(l0, fmaxf(l1, l2));
  const float e0 = __expf(l0 - lm), e1 = __expf(l1 - lm), e2 = __expf(l2 - lm);
  const float wsc = (scale_idx == 0 ? e0 : scale_idx == 1 ? e1 : e2) / (e0 + e1 + e2);

  // ---- pass 2: out += wsc * (P @ v_txt), A-fragments straight from LDS probs ----
  for (int cc = 0; cc < 8; ++cc) {
    v8f acc0 = zero_v8f(), acc1 = zero_v8f();
    const int cbase = cc * 128;
    for (int l0i = 0; l0i < L_DIM; l0i += 32) {
      __syncthreads();
      {  // v_txt tile 32(l) x 128(c) -> Bs[c][l]; pipeline both loads first
        uint4 dv[2];
#pragma unroll
        for (int j = 0; j < 2; ++j) {
          const int i = t + 256 * j;
          const int l = i >> 4, c8 = (i & 15) << 3;
          dv[j] = *(const uint4*)(vtxt + (size_t)(b * L_DIM + l0i + l) * C_DIM + cbase + c8);
        }
#pragma unroll
        for (int j = 0; j < 2; ++j) {
          const int i = t + 256 * j;
          const int l = i >> 4, c8 = (i & 15) << 3;
          const bf16_t* e = (const bf16_t*)&dv[j];
#pragma unroll
          for (int x = 0; x < 8; ++x) Bs[(c8 + x) * TSTR + l] = e[x];
        }
      }
      __syncthreads();
      const int arow = wm * 16 + lr;
      const int akb = (hi ? 8 : 0) + l0i;
      const v16bf af = cat8(*(const v8bf*)(P + arow * PSTR + akb),
                            *(const v8bf*)(P + arow * PSTR + akb + 16));
      const int bkb = hi ? 8 : 0;
      const int col0 = wn * 32 + lr, col1 = col0 + 16;
      const v16bf bf0 = cat8(*(const v8bf*)(Bs + col0 * TSTR + bkb),
                             *(const v8bf*)(Bs + col0 * TSTR + bkb + 16));
      const v16bf bf1 = cat8(*(const v8bf*)(Bs + col1 * TSTR + bkb),
                             *(const v8bf*)(Bs + col1 * TSTR + bkb + 16));
      acc0 = __builtin_amdgcn_wmma_f32_16x16x32_bf16(false, af, false, bf0, (short)0, acc0, false, false);
      acc1 = __builtin_amdgcn_wmma_f32_16x16x32_bf16(false, af, false, bf1, (short)0, acc1, false, false);
    }
#pragma unroll
    for (int v = 0; v < 8; ++v) {
      const int gn = nbase + wm * 16 + v + 8 * hi;
      const int gc0 = cbase + wn * 32 + lr;
      const size_t o0 = (size_t)b * C_DIM * N_DIM + (size_t)gc0 * N_DIM + gn;
      out[o0] += wsc * acc0[v];
      out[o0 + (size_t)16 * N_DIM] += wsc * acc1[v];
    }
  }
}

// ---------------- host ----------------
extern "C" void kernel_launch(void* const* d_in, const int* in_sizes, int n_in,
                              void* d_out, int out_size, void* d_ws, size_t ws_size,
                              hipStream_t stream) {
  const float* v_feat = (const float*)d_in[0];
  const float* t_tok  = (const float*)d_in[1];
  const float* Wq = (const float*)d_in[2];
  const float* bq = (const float*)d_in[3];
  const float* Wk = (const float*)d_in[4];
  const float* bk = (const float*)d_in[5];
  const float* Wv = (const float*)d_in[6];
  const float* bv = (const float*)d_in[7];
  const float* lam = (const float*)d_in[8];
  float* out = (float*)d_out;

  char* ws = (char*)d_ws;
  size_t off = 0;
  auto alloc = [&](size_t elems) -> bf16_t* {
    bf16_t* p = (bf16_t*)(ws + off);
    off = (off + elems * sizeof(bf16_t) + 255) & ~(size_t)255;
    return p;
  };
  bf16_t* tok_bf = alloc((size_t)B_DIM * L_DIM * CT_DIM);   // 6.29M
  bf16_t* wq_bf  = alloc((size_t)C_DIM * C_DIM);            // 1.05M
  bf16_t* wk_bf  = alloc((size_t)CT_DIM * C_DIM);
  bf16_t* wv_bf  = alloc((size_t)CT_DIM * C_DIM);
  bf16_t* ktxt   = alloc((size_t)B_DIM * L_DIM * C_DIM);    // 8.39M
  bf16_t* vtxt   = alloc((size_t)B_DIM * L_DIM * C_DIM);
  bf16_t* vflat  = alloc((size_t)B_DIM * N_DIM * C_DIM);    // 37.7M
  bf16_t* qbuf   = alloc((size_t)B_DIM * N_DIM * C_DIM);

  cvt_bf16_kernel<<<2048, 256, 0, stream>>>(t_tok, tok_bf, (size_t)B_DIM * L_DIM * CT_DIM);
  cvt_bf16_kernel<<<1024, 256, 0, stream>>>(Wq, wq_bf, (size_t)C_DIM * C_DIM);
  cvt_bf16_kernel<<<1024, 256, 0, stream>>>(Wk, wk_bf, (size_t)CT_DIM * C_DIM);
  cvt_bf16_kernel<<<1024, 256, 0, stream>>>(Wv, wv_bf, (size_t)CT_DIM * C_DIM);

  // k_txt / v_txt: (B*L=8192, C=1024, K=768)
  gemm_bias_bf16<<<dim3(C_DIM / 128, (B_DIM * L_DIM) / 128), 256, 0, stream>>>(
      tok_bf, wk_bf, bk, ktxt, B_DIM * L_DIM, C_DIM, CT_DIM);
  gemm_bias_bf16<<<dim3(C_DIM / 128, (B_DIM * L_DIM) / 128), 256, 0, stream>>>(
      tok_bf, wv_bf, bv, vtxt, B_DIM * L_DIM, C_DIM, CT_DIM);

  init_out_kernel<<<8192, 256, 0, stream>>>(v_feat, t_tok, out);

  const int scales[3] = {1, 3, 5};
  for (int idx = 0; idx < 3; ++idx) {
    pool_kernel<<<B_DIM * C_DIM, 256, 0, stream>>>(v_feat, vflat, scales[idx]);
    // q = v_flat @ Wq + bq : (B*N=36864, C=1024, K=1024)
    gemm_bias_bf16<<<dim3(C_DIM / 128, (B_DIM * N_DIM) / 128), 256, 0, stream>>>(
        vflat, wq_bf, bq, qbuf, B_DIM * N_DIM, C_DIM, C_DIM);
    attn_kernel<<<dim3(N_DIM / 32, B_DIM), 256, 0, stream>>>(qbuf, ktxt, vtxt, lam, out, idx);
  }
}
